// SimpleLSTM_33638183862581
// MI455X (gfx1250) — matrix operand order repaired and checked by reading, not compiled, not run on hardware
//
#include <hip/hip_runtime.h>
#include <hip/hip_bf16.h>
#include <math.h>

// ---------------- problem constants ----------------
#define TSEQ   512
#define BATCH  2048
#define HID    50
#define GATES  200          // 4*H
#define NTILE  13           // ceil(200/16)
#define GPITCH 224          // padded gate pitch in LDS
#define MT     16           // batch rows per block
#define KSMAX  4            // max k-steps of 32 (K = inDim + HID <= 100 -> 4)

typedef __bf16 bf16;
typedef bf16  v16bf __attribute__((ext_vector_type(16)));
typedef float v8f   __attribute__((ext_vector_type(8)));

union FragU { unsigned int u[8]; v16bf v; };

__device__ __forceinline__ unsigned int pack_bf2(float lo, float hi) {
  union { bf16 h[2]; unsigned int u; } p;
  p.h[0] = (bf16)lo; p.h[1] = (bf16)hi;
  return p.u;
}

// ---- A-matrix LDS layout: stored in exact WMMA fragment order -------------
// word index = (kstep*32 + lane)*8 + vgpr ; each word = 2 packed bf16.
// this code's contiguous k order (not ISA table order): lanes0-15 hold K 0..15,
// lanes16-31 hold K 16..31; 2 K-values per VGPR, low half first.
__device__ __forceinline__ void store_a_elem(unsigned int* aW, int m, int k, float val) {
  int kstep = k >> 5;
  int kk    = k & 31;
  int grp   = kk >> 3;                      // 0..3
  int lane  = m + ((kk >> 4) << 4);
  int v     = (kk & 15) >> 1;
  int hw    = kk & 1;
  unsigned short* hp = (unsigned short*)aW;
  union { bf16 b; unsigned short s; } c; c.b = (bf16)val;
  hp[(((kstep << 5) + lane) * 8 + v) * 2 + hw] = c.s;
}

__device__ __forceinline__ v16bf load_a_frag(const unsigned int* aW, int kstep, int lane) {
  FragU f;
  const unsigned int* p = aW + (kstep * 32 + lane) * 8;   // 32B contiguous per lane
#pragma unroll
  for (int i = 0; i < 8; ++i) f.u[i] = p[i];
  return f.v;
}

// fused weight element: column k of [W_ih ; W_hh] row g (zero-padded)
__device__ __forceinline__ float wAt(const float* __restrict__ wih,
                                     const float* __restrict__ whh,
                                     int inDim, int g, int k) {
  if (g >= GATES) return 0.f;
  if (k < inDim)  return wih[g * inDim + k];
  int kh = k - inDim;
  if (kh < HID)   return whh[g * HID + kh];
  return 0.f;
}

// B-matrix 32x16 bf16 fragment (contiguous k, same as A: lanes0-15 -> K 0..15, lanes16-31 -> K 16..31,
// n = lane%16, 2 K per VGPR, low half first)
__device__ __forceinline__ v16bf load_b_frag(const float* __restrict__ wih,
                                             const float* __restrict__ whh,
                                             int inDim, int tile, int kstep, int lane) {
  int n  = lane & 15;
  int g  = tile * 16 + n;
  int kb = kstep * 32 + ((lane >> 4) << 4);
  FragU f;
#pragma unroll
  for (int v = 0; v < 8; ++v) {
    float a = wAt(wih, whh, inDim, g, kb + 2 * v);
    float b = wAt(wih, whh, inDim, g, kb + 2 * v + 1);
    f.u[v] = pack_bf2(a, b);
  }
  return f.v;
}

__device__ __forceinline__ v16bf zero_frag() {
  FragU f;
#pragma unroll
  for (int i = 0; i < 8; ++i) f.u[i] = 0u;
  return f.v;
}

__device__ __forceinline__ float sigmoidf_(float x) { return 1.f / (1.f + expf(-x)); }

// ---------------- fused LSTM layer kernel ----------------
// One block = 16 batch rows, loops all T timesteps.
// gates = [x_t , h_{t-1}] @ [W_ih ; W_hh]^T + (b_ih + b_hh) via bf16 WMMA, f32 acc.
template <int IND, bool FIRST>
__global__ __launch_bounds__(256)
void lstm_layer(const float* __restrict__ xin,      // layer0 input  [B,T,IND] f32
                const bf16*  __restrict__ hin,      // layer>0 input [B,T,IND] bf16
                bf16*        __restrict__ hout,     // output        [B,T,HID] bf16
                const float* __restrict__ w_ih,     // [GATES, IND]
                const float* __restrict__ w_hh,     // [GATES, HID]
                const float* __restrict__ b_ih,     // [GATES]
                const float* __restrict__ b_hh)     // [GATES]
{
  constexpr int NK = (IND + HID + 31) / 32;         // 3 (layer0) or 4

  __shared__ unsigned int aW[KSMAX * 32 * 8];       // A fragments, 4 KB
  __shared__ float gatesS[MT * GPITCH];             // 14 KB
  __shared__ float bv[GPITCH];                      // fused bias
  __shared__ float cs[MT * HID];                    // cell state, f32

  const int tid   = threadIdx.x;
  const int lane  = tid & 31;
  const int bBase = blockIdx.x * MT;

  // Scalar (SGPR) wave id: the wave index is uniform within a wave, so
  // readfirstlane makes the tile-1 branch a uniform s_cbranch -- WMMA then
  // always executes with EXEC all-1s (ISA 7.12 requirement), with no
  // s_and_saveexec predication in the 512-iteration hot loop.
  const int  wave_u = __builtin_amdgcn_readfirstlane(tid >> 5);
  const int  tile0  = wave_u;              // tiles 0..7
  const int  tile1  = wave_u + 8;          // tiles 8..12 on waves 0..4
  const bool has1   = (tile1 < NTILE);

  // ---- init LDS ----
  for (int i = tid; i < KSMAX * 32 * 8; i += 256) aW[i] = 0u;     // zero pads & h(t=-1)=0
  for (int i = tid; i < MT * HID;       i += 256) cs[i] = 0.f;
  if (tid < GPITCH) bv[tid] = (tid < GATES) ? (b_ih[tid] + b_hh[tid]) : 0.f;

  // ---- resident B-matrix fragments (weights), loaded once ----
  v16bf bf0[KSMAX], bf1[KSMAX];
#pragma unroll
  for (int s = 0; s < KSMAX; ++s) {
    bf0[s] = (s < NK) ? load_b_frag(w_ih, w_hh, IND, tile0, s, lane) : zero_frag();
    bf1[s] = (s < NK && has1) ? load_b_frag(w_ih, w_hh, IND, tile1, s, lane) : zero_frag();
  }

  // ---- preload input for t = 0 ----
  for (int p = tid; p < MT * IND; p += 256) {
    int m = p / IND, d = p - m * IND;
    size_t idx = ((size_t)(bBase + m) * TSEQ + 0) * IND + d;
    float v = FIRST ? xin[idx] : (float)hin[idx];
    store_a_elem(aW, m, d, v);
  }
  __syncthreads();

  // ---- timestep loop ----
  for (int t = 0; t < TSEQ; ++t) {
    // GEMM: uniform control flow, EXEC all ones -> v_wmma_f32_16x16x32_bf16
    v8f acc0 = {}; v8f acc1 = {};
#pragma unroll
    for (int s = 0; s < NK; ++s) {
      v16bf a = load_a_frag(aW, s, lane);
      acc0 = __builtin_amdgcn_wmma_f32_16x16x32_bf16(false, a, false, bf0[s],
                                                     (short)0, acc0, false, false);
      if (has1)   // scalar branch (SGPR condition)
        acc1 = __builtin_amdgcn_wmma_f32_16x16x32_bf16(false, a, false, bf1[s],
                                                       (short)0, acc1, false, false);
    }
    // C layout: VGPR r, lane L -> m = r + 8*(L/16), n = L%16
    {
      const int mh = (lane >> 4) << 3;
      const int n0 = lane & 15;
#pragma unroll
      for (int r = 0; r < 8; ++r)
        gatesS[(mh + r) * GPITCH + tile0 * 16 + n0] = acc0[r];
      if (has1) {
#pragma unroll
        for (int r = 0; r < 8; ++r)
          gatesS[(mh + r) * GPITCH + tile1 * 16 + n0] = acc1[r];
      }
    }
    __syncthreads();

    // elementwise LSTM cell; writes h into A-LDS (k = IND + j) for next step
    for (int p = tid; p < MT * HID; p += 256) {
      int m = p / HID, j = p - m * HID;
      float gi = gatesS[m * GPITCH + j]             + bv[j];
      float gf = gatesS[m * GPITCH + HID + j]       + bv[HID + j];
      float gg = gatesS[m * GPITCH + 2 * HID + j]   + bv[2 * HID + j];
      float go = gatesS[m * GPITCH + 3 * HID + j]   + bv[3 * HID + j];
      float i_ = sigmoidf_(gi);
      float f_ = sigmoidf_(gf);
      float g_ = tanhf(gg);
      float o_ = sigmoidf_(go);
      float c  = f_ * cs[p] + i_ * g_;
      cs[p] = c;
      float h = o_ * tanhf(c);
      hout[((size_t)(bBase + m) * TSEQ + t) * HID + j] = (bf16)h;
      store_a_elem(aW, m, IND + j, h);
    }

    // stage next timestep's input into A-LDS (k = 0..IND-1)
    if (t + 1 < TSEQ) {
      for (int p = tid; p < MT * IND; p += 256) {
        int m = p / IND, d = p - m * IND;
        size_t idx = ((size_t)(bBase + m) * TSEQ + (t + 1)) * IND + d;
        float v = FIRST ? xin[idx] : (float)hin[idx];
        store_a_elem(aW, m, d, v);
      }
      if (t + 2 < TSEQ && tid < MT) {               // gfx1250 global_prefetch_b8
        size_t idx2 = ((size_t)(bBase + tid) * TSEQ + (t + 2)) * IND;
        if (FIRST) __builtin_prefetch((const void*)(xin + idx2), 0, 1);
        else       __builtin_prefetch((const void*)(hin + idx2), 0, 1);
      }
    }
    __syncthreads();
  }
}

// ---------------- MLP head: gelu(h_last @ w1^T + b1) @ w2^T + b2 ----------------
__global__ __launch_bounds__(256)
void head_kernel(const bf16* __restrict__ hbuf,
                 const float* __restrict__ w1, const float* __restrict__ b1,
                 const float* __restrict__ w2, const float* __restrict__ b2,
                 float* __restrict__ out)
{
  int b = blockIdx.x * blockDim.x + threadIdx.x;
  if (b >= BATCH) return;
  float h[HID];
  const bf16* hp = hbuf + ((size_t)b * TSEQ + (TSEQ - 1)) * HID;
#pragma unroll
  for (int j = 0; j < HID; ++j) h[j] = (float)hp[j];
  float y = b2[0];
#pragma unroll 5
  for (int i = 0; i < 25; ++i) {
    float s = b1[i];
    for (int j = 0; j < HID; ++j) s += h[j] * w1[i * HID + j];
    float g = 0.5f * s * (1.f + erff(s * 0.70710678118654752f));  // exact GELU
    y += g * w2[i];
  }
  out[b] = y;
}

// ---------------- host side ----------------
extern "C" void kernel_launch(void* const* d_in, const int* in_sizes, int n_in,
                              void* d_out, int out_size, void* d_ws, size_t ws_size,
                              hipStream_t stream) {
  (void)in_sizes; (void)n_in; (void)out_size;
  const float* x     = (const float*)d_in[0];
  const float* w_ih0 = (const float*)d_in[1];
  const float* w_ih  = (const float*)d_in[2];   // [7,200,50]
  const float* w_hh  = (const float*)d_in[3];   // [8,200,50]
  const float* b_ih  = (const float*)d_in[4];   // [8,200]
  const float* b_hh  = (const float*)d_in[5];   // [8,200]
  const float* w1    = (const float*)d_in[6];
  const float* b1    = (const float*)d_in[7];
  const float* w2    = (const float*)d_in[8];
  const float* b2    = (const float*)d_in[9];

  const size_t bufBytes = (size_t)BATCH * TSEQ * HID * sizeof(bf16);  // ~100 MB
  bf16* bufA = (bf16*)d_ws;
  const bool pp = (ws_size >= 2 * bufBytes);
  bf16* bufB = pp ? (bf16*)((char*)d_ws + bufBytes) : bufA;  // in-place fallback is safe:
  // each block only reads (b, t+1) of its own batch rows, which it has not yet written.

  dim3 grid(BATCH / MT), block(256);

  lstm_layer<18, true><<<grid, block, 0, stream>>>(
      x, (const bf16*)nullptr, bufA, w_ih0, w_hh, b_ih, b_hh);

  bf16* cur = bufA; bf16* nxt = bufB;
  for (int l = 1; l < 8; ++l) {
    lstm_layer<HID, false><<<grid, block, 0, stream>>>(
        (const float*)nullptr, cur, nxt,
        w_ih + (size_t)(l - 1) * GATES * HID,
        w_hh + (size_t)l * GATES * HID,
        b_ih + (size_t)l * GATES,
        b_hh + (size_t)l * GATES);
    bf16* tmp = cur; cur = nxt; nxt = tmp;
  }

  head_kernel<<<dim3(BATCH / 256), block, 0, stream>>>(cur, w1, b1, w2, b2, (float*)d_out);
}
